// MOE_FFN_38800734552690
// MI455X (gfx1250) — matrix operand order repaired and checked by reading, not compiled, not run on hardware
//
#include <hip/hip_runtime.h>
#include <math.h>

// Problem constants (from reference): B=4, S=1024, D=1024, E=8, K=2, NS=2, H=4096
#define T_TOKENS 4096
#define DDIM     1024
#define HDIM     4096
#define NEXP     8
#define NSHARED  2
#define TEMP_INV 10.0f

typedef __bf16 bf16;
typedef __attribute__((ext_vector_type(8)))  __bf16 v8bf;
typedef __attribute__((ext_vector_type(16))) __bf16 v16bf;
typedef __attribute__((ext_vector_type(8)))  float  v8f;
typedef __attribute__((ext_vector_type(4)))  unsigned int u32x4;
typedef __attribute__((ext_vector_type(8)))  unsigned int u32x8;

// ---------------- small utility kernels ----------------

__global__ void zero_f32(float* __restrict__ p, int n) {
    int i = blockIdx.x * 256 + threadIdx.x;
    if (i < n) p[i] = 0.0f;
}

__global__ void cvt_bf16(const float* __restrict__ x, bf16* __restrict__ y, int n) {
    int i = blockIdx.x * 256 + threadIdx.x;
    if (i < n) y[i] = (bf16)x[i];
}

// Convert + transpose weights: W f32 [K][N] (row-major) -> Wt bf16 [N][K].
// Makes the GEMM's B tile a straight contiguous-K copy (TDM-friendly).
__global__ void __launch_bounds__(256)
cvt_w_t(const float* __restrict__ W, bf16* __restrict__ Wt, int K, int N) {
    __shared__ float tile[64][68];          // 272B row stride (16B aligned)
    const int tid   = threadIdx.x;
    const int nBase = blockIdx.x * 64;
    const int kBase = blockIdx.y * 64;
    #pragma unroll
    for (int i = 0; i < 4; ++i) {
        int idx = i * 256 + tid;            // 1024 float4 chunks
        int kr = idx >> 4, nc = (idx & 15) * 4;
        float4 v = *(const float4*)(W + (size_t)(kBase + kr) * N + nBase + nc);
        *(float4*)&tile[kr][nc] = v;
    }
    __syncthreads();
    #pragma unroll
    for (int i = 0; i < 2; ++i) {
        int idx = i * 256 + tid;            // 512 chunks of 8 bf16
        int nr = idx >> 3, kc = (idx & 7) * 8;
        v8bf o;
        #pragma unroll
        for (int j = 0; j < 8; ++j) o[j] = (bf16)tile[kc + j][nr];
        *(v8bf*)(Wt + (size_t)(nBase + nr) * K + kBase + kc) = o;
    }
}

// Router: mask[t][e] = 1 if e in top-2 of (x@Wr + br + gumbel)/TEMP
__global__ void __launch_bounds__(256)
router_kernel(const float* __restrict__ x, const float* __restrict__ Wr,
              const float* __restrict__ br, const float* __restrict__ gum,
              float* __restrict__ mask) {
    __shared__ float z[32][NEXP];
    const int tid = threadIdx.x;
    const int tl  = tid >> 3;
    const int e   = tid & 7;
    const int t   = blockIdx.x * 32 + tl;
    const float* xr = x + (size_t)t * DDIM;
    float acc = 0.0f;
    for (int d = 0; d < DDIM; ++d) acc += xr[d] * Wr[d * NEXP + e];
    z[tl][e] = (acc + br[e] + gum[(size_t)t * NEXP + e]) * TEMP_INV;
    __syncthreads();
    if (e == 0) {
        int i1 = 0; float m1 = z[tl][0];
        for (int j = 1; j < NEXP; ++j) if (z[tl][j] > m1) { m1 = z[tl][j]; i1 = j; }
        int i2 = -1; float m2 = -1e30f;
        for (int j = 0; j < NEXP; ++j) if (j != i1 && z[tl][j] > m2) { m2 = z[tl][j]; i2 = j; }
        for (int j = 0; j < NEXP; ++j)
            mask[(size_t)t * NEXP + j] = (j == i1 || j == i2) ? 1.0f : 0.0f;
    }
}

// ---------------- Tensor Data Mover tile load ----------------
// 2D tile: 128 rows x 32 contiguous-K bf16 elements, from a [rows_total][K]
// bf16 tensor, into LDS with row padding 32 -> 40 elements (pad 16B every 64B:
// pad_interval code 3 = 16 DWORDs, pad_amount code 3 = 4 DWORDs).
__device__ __forceinline__ void tdm_load_tile(unsigned lds_off, const bf16* gptr,
                                              unsigned Klen, unsigned rows_total) {
    unsigned long long ga = (unsigned long long)gptr;
    u32x4 g0;
    g0.x = 1u;                                          // count=1 (valid user D#)
    g0.y = lds_off;                                     // lds_addr (bytes)
    g0.z = (unsigned)(ga & 0xFFFFFFFFu);                // global_addr[31:0]
    g0.w = (unsigned)((ga >> 32) & 0x1FFFFFFu)          // global_addr[56:32]
         | (2u << 30);                                  // type=2 ("image")
    u32x8 g1;
    g1.s0 = (1u << 16)                                  // data_size=1 (2 bytes)
          | (1u << 20) | (3u << 22) | (3u << 25);       // pad_en, interval=16dw, amount=4dw
    g1.s1 = (Klen & 0xFFFFu) << 16;                     // tensor_dim0[15:0]
    g1.s2 = (Klen >> 16) | ((rows_total & 0xFFFFu) << 16); // dim0[31:16] | dim1[15:0]
    g1.s3 = (rows_total >> 16) | (32u << 16);           // dim1[31:16] | tile_dim0=32
    g1.s4 = 128u;                                       // tile_dim1=128, tile_dim2=0
    g1.s5 = Klen;                                       // tensor_dim0_stride[31:0]
    g1.s6 = 0u;                                         // stride0[47:32] | stride1[15:0]
    g1.s7 = 0u;                                         // stride1[47:16]
    asm volatile("tensor_load_to_lds %0, %1" :: "s"(g0), "s"(g1) : "memory");
}

// ---------------- WMMA GEMM ----------------
// C[M,N] = A[M,K](bf16 row-major) x B, with B given TRANSPOSED: Bt[N][K] bf16.
// MODE 0: Out(bf16) = gelu(C + bias[n])
// MODE 1: Out(f32) += scale(row) * (C + bias[n])

#define BM 128
#define BN 128
#define BK 32
#define LDSA 40   // padded row stride (elements) produced by TDM pad config

__device__ __forceinline__ float gelu_f(float x) {
    return 0.5f * x * (1.0f + tanhf(0.7978845608028654f * (x + 0.044715f * x * x * x)));
}

template <int MODE>
__global__ void __launch_bounds__(256)
ffn_gemm(const bf16* __restrict__ A, const bf16* __restrict__ BtG,
         const float* __restrict__ bias, void* __restrict__ OutP,
         const float* __restrict__ maskCol, int eIdx, int N, int K) {
    __shared__ bf16 As[2][BM][LDSA];   // double-buffered A tile [m][k]
    __shared__ bf16 Bs[2][BN][LDSA];   // double-buffered B tile [n][k]

    const int tid   = threadIdx.x;
    const int lane  = tid & 31;
    const int wave  = tid >> 5;
    const int wm    = wave & 3;        // 4 wave rows x 32
    const int wn    = wave >> 2;       // 2 wave cols x 64
    const int half  = lane >> 4;
    const int lrow  = lane & 15;
    const int mBase = blockIdx.y * BM;
    const int nBase = blockIdx.x * BN;

    const bf16* aTile = A   + (size_t)mBase * K;
    const bf16* bTile = BtG + (size_t)nBase * K;
    const unsigned ldsA0 = (unsigned)(unsigned long long)(void*)&As[0][0][0];
    const unsigned ldsA1 = (unsigned)(unsigned long long)(void*)&As[1][0][0];
    const unsigned ldsB0 = (unsigned)(unsigned long long)(void*)&Bs[0][0][0];
    const unsigned ldsB1 = (unsigned)(unsigned long long)(void*)&Bs[1][0][0];

    v8f acc[2][4];
    const v8f zf = {0.f, 0.f, 0.f, 0.f, 0.f, 0.f, 0.f, 0.f};
    #pragma unroll
    for (int i = 0; i < 2; ++i)
        #pragma unroll
        for (int j = 0; j < 4; ++j) acc[i][j] = zf;

    const int nk = K / BK;
    if (wave == 0) {                   // prime the pipeline: tile 0 -> buffer 0
        tdm_load_tile(ldsA0, aTile, (unsigned)K, (unsigned)T_TOKENS);
        tdm_load_tile(ldsB0, bTile, (unsigned)K, (unsigned)N);
    }

    for (int kt = 0; kt < nk; ++kt) {
        const int cur = kt & 1;
        __syncthreads();               // next-buffer LDS free; prev compute done
        if (wave == 0) {
            if (kt + 1 < nk) {         // issue next tile into the other buffer
                const int k1 = (kt + 1) * BK;
                tdm_load_tile(cur ? ldsA0 : ldsA1, aTile + k1, (unsigned)K, (unsigned)T_TOKENS);
                tdm_load_tile(cur ? ldsB0 : ldsB1, bTile + k1, (unsigned)K, (unsigned)N);
                __builtin_amdgcn_s_wait_tensorcnt(2);   // current pair complete (in-order)
            } else {
                __builtin_amdgcn_s_wait_tensorcnt(0);
            }
        }
        __syncthreads();               // current tile visible to all waves

        // B fragments: lane = column n (lrow); K 0..15 (lanes<16) / 16..31
        v16bf bfr[4];
        #pragma unroll
        for (int nt = 0; nt < 4; ++nt) {
            int bc = wn * 64 + nt * 16 + lrow;
            v8bf b0 = *(const v8bf*)&Bs[cur][bc][half * 16];
            v8bf b1 = *(const v8bf*)&Bs[cur][bc][half * 16 + 8];
            bfr[nt] = __builtin_shufflevector(b0, b1, 0, 1, 2, 3, 4, 5, 6, 7,
                                              8, 9, 10, 11, 12, 13, 14, 15);
        }
        // A fragments: lane = row m (lrow); K 0..15 (lanes<16) / 16..31
        #pragma unroll
        for (int mt = 0; mt < 2; ++mt) {
            int ar = wm * 32 + mt * 16 + lrow;
            v8bf a0 = *(const v8bf*)&As[cur][ar][half * 16];
            v8bf a1 = *(const v8bf*)&As[cur][ar][half * 16 + 8];
            v16bf af = __builtin_shufflevector(a0, a1, 0, 1, 2, 3, 4, 5, 6, 7,
                                               8, 9, 10, 11, 12, 13, 14, 15);
            #pragma unroll
            for (int nt = 0; nt < 4; ++nt) {
                acc[mt][nt] = __builtin_amdgcn_wmma_f32_16x16x32_bf16(
                    false, af, false, bfr[nt], (short)0, acc[mt][nt], false, false);
            }
        }
    }

    // epilogue: C VGPR i -> row = i + 8*half, col = lrow (ISA C/D layout)
    #pragma unroll
    for (int mt = 0; mt < 2; ++mt) {
        #pragma unroll
        for (int nt = 0; nt < 4; ++nt) {
            const int col = nBase + wn * 64 + nt * 16 + lrow;
            const float b = bias[col];
            #pragma unroll
            for (int i = 0; i < 8; ++i) {
                const int row = mBase + wm * 32 + mt * 16 + half * 8 + i;
                const float v = acc[mt][nt][i] + b;
                if (MODE == 0) {
                    bf16* Out = (bf16*)OutP;
                    Out[(size_t)row * N + col] = (bf16)gelu_f(v);
                } else {
                    float* Out = (float*)OutP;
                    const float sc = maskCol ? maskCol[(size_t)row * NEXP + eIdx] : 1.0f;
                    Out[(size_t)row * N + col] += sc * v;
                }
            }
        }
    }
}

// ---------------- launcher ----------------

extern "C" void kernel_launch(void* const* d_in, const int* in_sizes, int n_in,
                              void* d_out, int out_size, void* d_ws, size_t ws_size,
                              hipStream_t stream) {
    const float* x   = (const float*)d_in[0];
    const float* Ws1 = (const float*)d_in[1];
    const float* bs1 = (const float*)d_in[2];
    const float* Ws2 = (const float*)d_in[3];
    const float* bs2 = (const float*)d_in[4];
    const float* We1 = (const float*)d_in[5];
    const float* be1 = (const float*)d_in[6];
    const float* We2 = (const float*)d_in[7];
    const float* be2 = (const float*)d_in[8];
    const float* Wr  = (const float*)d_in[9];
    const float* br  = (const float*)d_in[10];
    const float* gum = (const float*)d_in[11];
    float* out = (float*)d_out;

    // workspace layout (bf16 staging buffers, reused across experts)
    char* ws = (char*)d_ws;
    size_t off = 0;
    bf16* xb  = (bf16*)(ws + off); off += (size_t)T_TOKENS * DDIM * sizeof(bf16); //  8 MB
    bf16* hb  = (bf16*)(ws + off); off += (size_t)T_TOKENS * HDIM * sizeof(bf16); // 32 MB
    bf16* wt1 = (bf16*)(ws + off); off += (size_t)DDIM * HDIM * sizeof(bf16);     //  8 MB  Wt1[H][D]
    bf16* wt2 = (bf16*)(ws + off); off += (size_t)HDIM * DDIM * sizeof(bf16);     //  8 MB  Wt2[D][H]
    float* mask = (float*)(ws + off);                                             // 128 KB

    const int nOut = T_TOKENS * DDIM;
    zero_f32<<<(nOut + 255) / 256, 256, 0, stream>>>(out, nOut);
    cvt_bf16<<<(nOut + 255) / 256, 256, 0, stream>>>(x, xb, nOut);
    router_kernel<<<T_TOKENS / 32, 256, 0, stream>>>(x, Wr, br, gum, mask);

    dim3 g1(HDIM / BN, T_TOKENS / BM);   // stage-1 GEMM: N=H, K=D
    dim3 g2(DDIM / BN, T_TOKENS / BM);   // stage-2 GEMM: N=D, K=H
    dim3 gw1(HDIM / 64, DDIM / 64);      // W1 [D][H] -> Wt1 [H][D]
    dim3 gw2(DDIM / 64, HDIM / 64);      // W2 [H][D] -> Wt2 [D][H]

    for (int s = 0; s < NSHARED; ++s) {
        cvt_w_t<<<gw1, 256, 0, stream>>>(Ws1 + (size_t)s * DDIM * HDIM, wt1, DDIM, HDIM);
        ffn_gemm<0><<<g1, 256, 0, stream>>>(xb, wt1, bs1 + (size_t)s * HDIM, hb,
                                            nullptr, 0, HDIM, DDIM);
        cvt_w_t<<<gw2, 256, 0, stream>>>(Ws2 + (size_t)s * HDIM * DDIM, wt2, HDIM, DDIM);
        ffn_gemm<1><<<g2, 256, 0, stream>>>(hb, wt2, bs2 + (size_t)s * DDIM, out,
                                            nullptr, 0, DDIM, HDIM);
    }
    for (int e = 0; e < NEXP; ++e) {
        cvt_w_t<<<gw1, 256, 0, stream>>>(We1 + (size_t)e * DDIM * HDIM, wt1, DDIM, HDIM);
        ffn_gemm<0><<<g1, 256, 0, stream>>>(xb, wt1, be1 + (size_t)e * HDIM, hb,
                                            nullptr, 0, HDIM, DDIM);
        cvt_w_t<<<gw2, 256, 0, stream>>>(We2 + (size_t)e * HDIM * DDIM, wt2, HDIM, DDIM);
        ffn_gemm<1><<<g2, 256, 0, stream>>>(hb, wt2, be2 + (size_t)e * DDIM, out,
                                            mask, e, DDIM, HDIM);
    }
}
